// MultiTaskHockeyGNN_24696061952698
// MI455X (gfx1250) — matrix operand order repaired and checked by reading, not compiled, not run on hardware
//
#include <hip/hip_runtime.h>
#include <hip/hip_bf16.h>
#include <math.h>

// ---------------------------------------------------------------------------
// MultiTaskHockeyGNN forward for gfx1250 (MI455X, wave32, WMMA)
//
//   1. deg/dis:   dis[i] = rsqrt(#edges_into_i + 1)          (self loop => deg>=1)
//   2. h  = relu(bn(x @ W_in + b))                           [WMMA bf16 GEMM]
//   3. 3x GCN:  hw = h @ W                                   [WMMA bf16 GEMM]
//               acc = scatter_add(norm_e * hw[src] -> dst)   [f32 atomics]
//               h  += relu(bn(acc + dis^2*hw + b))           (self loop fused)
//   4. g = h[game_indices]
//   5. 5 heads: WMMA GEMMs (bias/BN/ReLU fused epilogue) + tiny 2-logit
//      log-softmax kernel.
//
// B matrices are pre-swizzled into exact WMMA fragment order so each lane
// fetches its fragment with one contiguous 32B global load (weights are tiny
// and L2-resident; no LDS staging / barriers needed). A rows are clamped (not
// masked) at the M edge: garbage A rows only affect C rows that the epilogue
// never stores, so the inner loop stays branch-free.
// ---------------------------------------------------------------------------

typedef __attribute__((ext_vector_type(16))) __bf16 v16bf;
typedef __attribute__((ext_vector_type(8)))  float  v8f;

#define EP_BIAS 1
#define EP_BN   2
#define EP_RELU 4
#define BN_EPS  1e-5f

// ---------------------------------------------------------------------------
// Pack W (f32, K x N row-major) into bf16 WMMA B fragments:
//   Bp[tile*512 + lane*16 + j] = W[kt*32 + (j>>3)*16 + (lane>>4)*8 + (j&7)][nt*16 + (lane&15)]
//   tile = kt*(N/16) + nt
// ---------------------------------------------------------------------------
__global__ void pack_b(const float* __restrict__ W, __bf16* __restrict__ Bp,
                       int K, int N) {
  int t = blockIdx.x * blockDim.x + threadIdx.x;
  if (t >= K * N) return;
  int j    = t & 15;
  int lane = (t >> 4) & 31;
  int tile = t >> 9;
  int ntiles = N >> 4;
  int kt = tile / ntiles, nt = tile - kt * ntiles;
  int k = kt * 32 + (j >> 3) * 16 + (lane >> 4) * 8 + (j & 7);
  int n = nt * 16 + (lane & 15);
  Bp[t] = (__bf16)W[(size_t)k * N + n];
}

// ---------------------------------------------------------------------------
// WMMA GEMM: out[M,N] = epilogue(A[M,K] @ B[K,N])
// Block = 256 threads = 8 waves. Each wave computes a 16x32 C strip (two
// 16x16 tiles sharing one A fragment -> 2 WMMAs/step, half the A traffic).
// Grid = (ceil(M/128), N/32).  K % 32 == 0, N % 32 == 0 for all uses here.
// ---------------------------------------------------------------------------
__global__ void gemm_wmma_bf16(
    const float* __restrict__ A, const __bf16* __restrict__ Bp,
    const float* __restrict__ bias,
    const float* __restrict__ bnG, const float* __restrict__ bnB,
    const float* __restrict__ bnM, const float* __restrict__ bnV,
    float* __restrict__ out, int M, int K, int N, int flags)
{
  const int wave    = threadIdx.x >> 5;   // 0..7
  const int lane    = threadIdx.x & 31;
  const int l15     = lane & 15;
  const int halfsel = lane >> 4;          // 0 or 1
  const int rowTile = blockIdx.x * 128 + wave * 16;
  const int colTile = blockIdx.y * 32;    // two 16-wide output tiles
  const int row     = rowTile + l15;
  const int kbaseA  = halfsel * 8;        // A layout: lanes>=16 hold K+8 block
  const int ntiles  = N >> 4;
  const int nt0     = blockIdx.y * 2;

  // Clamp instead of mask: out-of-range rows feed only C rows we never store.
  const int rowC = (row < M) ? row : (M - 1);

  const float*  ap  = A + (size_t)rowC * K + kbaseA;
  const __bf16* bp0 = Bp + (size_t)nt0 * 512 + (size_t)lane * 16;
  const size_t  bstep = (size_t)ntiles * 512;   // advance one k-tile

  v8f c0 = {}, c1 = {};

  for (int k0 = 0; k0 < K; k0 += 32) {
    // A fragment: lane l15 -> row, K = {k0+kbaseA..+7, k0+16+kbaseA..+7}
    const float4 p0 = *(const float4*)(ap + k0);
    const float4 p1 = *(const float4*)(ap + k0 + 4);
    const float4 p2 = *(const float4*)(ap + k0 + 16);
    const float4 p3 = *(const float4*)(ap + k0 + 20);
    v16bf a;
    a[0]  = (__bf16)p0.x; a[1]  = (__bf16)p0.y; a[2]  = (__bf16)p0.z; a[3]  = (__bf16)p0.w;
    a[4]  = (__bf16)p1.x; a[5]  = (__bf16)p1.y; a[6]  = (__bf16)p1.z; a[7]  = (__bf16)p1.w;
    a[8]  = (__bf16)p2.x; a[9]  = (__bf16)p2.y; a[10] = (__bf16)p2.z; a[11] = (__bf16)p2.w;
    a[12] = (__bf16)p3.x; a[13] = (__bf16)p3.y; a[14] = (__bf16)p3.z; a[15] = (__bf16)p3.w;

    // Two pre-swizzled B fragments (adjacent n-tiles), 32B contiguous each
    v16bf b0 = *(const v16bf*)(bp0);
    v16bf b1 = *(const v16bf*)(bp0 + 512);
    bp0 += bstep;

    c0 = __builtin_amdgcn_wmma_f32_16x16x32_bf16(
             false, a, false, b0, (short)0, c0, false, false);
    c1 = __builtin_amdgcn_wmma_f32_16x16x32_bf16(
             false, a, false, b1, (short)0, c1, false, false);
  }

  // Epilogue. C layout: c[v] -> row = rowTile + v + 8*halfsel, col = tile+l15
#pragma unroll
  for (int t = 0; t < 2; ++t) {
    const v8f& c = t ? c1 : c0;
    const int col = colTile + t * 16 + l15;
    float bv = (flags & EP_BIAS) ? bias[col] : 0.0f;
    float gm = 1.0f, bt = 0.0f, mn = 0.0f, inv = 1.0f;
    if (flags & EP_BN) {
      gm = bnG[col]; bt = bnB[col]; mn = bnM[col];
      inv = rsqrtf(bnV[col] + BN_EPS);
    }
#pragma unroll
    for (int v = 0; v < 8; ++v) {
      int r = rowTile + halfsel * 8 + v;
      if (r < M) {
        float val = c[v] + bv;
        if (flags & EP_BN)   val = (val - mn) * inv * gm + bt;
        if (flags & EP_RELU) val = fmaxf(val, 0.0f);
        out[(size_t)r * N + col] = val;
      }
    }
  }
}

// ---------------------------------------------------------------------------
__global__ void zero_f32(float* __restrict__ p, long n) {
  long i = (long)blockIdx.x * blockDim.x + threadIdx.x;
  if (i < n) p[i] = 0.0f;
}

__global__ void deg_accum(const int* __restrict__ dst, float* __restrict__ deg, int E) {
  int e = blockIdx.x * blockDim.x + threadIdx.x;
  if (e < E) atomicAdd(&deg[dst[e]], 1.0f);
}

__global__ void dis_finalize(float* __restrict__ dis, int n) {
  int i = blockIdx.x * blockDim.x + threadIdx.x;
  if (i < n) dis[i] = rsqrtf(dis[i] + 1.0f);   // +1 for self loop; deg >= 1 always
}

// 32 lanes per edge, 4 channels per lane (float4 gather, 4 f32 atomics)
__global__ void edge_scatter(const float* __restrict__ hw, const int* __restrict__ src,
                             const int* __restrict__ dst, const float* __restrict__ dis,
                             float* __restrict__ acc, int E) {
  long t = (long)blockIdx.x * blockDim.x + threadIdx.x;
  int e = (int)(t >> 5);
  if (e >= E) return;
  int c = ((int)t & 31) * 4;
  int s = src[e], d = dst[e];
  float nm = dis[s] * dis[d];
  const float4 v = *(const float4*)(hw + (size_t)s * 128 + c);
  float* a = acc + (size_t)d * 128 + c;
  atomicAdd(a + 0, nm * v.x);
  atomicAdd(a + 1, nm * v.y);
  atomicAdd(a + 2, nm * v.z);
  atomicAdd(a + 3, nm * v.w);
}

// h += relu(bn(acc + dis^2*hw + b))   (self-loop term fused, no atomics)
__global__ void gcn_finalize(float* __restrict__ h, const float* __restrict__ hw,
                             const float* __restrict__ acc, const float* __restrict__ dis,
                             const float* __restrict__ b,
                             const float* __restrict__ gamma, const float* __restrict__ beta,
                             const float* __restrict__ mean,  const float* __restrict__ var,
                             int nNodes) {
  long t = (long)blockIdx.x * blockDim.x + threadIdx.x;
  if (t >= (long)nNodes * 128) return;
  int i = (int)(t >> 7), c = (int)t & 127;
  float di = dis[i];
  float val = acc[t] + di * di * hw[t] + b[c];
  val = (val - mean[c]) * rsqrtf(var[c] + BN_EPS) * gamma[c] + beta[c];
  h[t] += fmaxf(val, 0.0f);
}

__global__ void gather_rows(const float* __restrict__ h, const int* __restrict__ idx,
                            float* __restrict__ g, int G) {
  long t = (long)blockIdx.x * blockDim.x + threadIdx.x;
  if (t >= (long)G * 128) return;
  int gi = (int)(t >> 7), c = (int)t & 127;
  g[t] = h[(size_t)idx[gi] * 128 + c];
}

// final Linear(D,2) + log_softmax, one thread per game
__global__ void head_final(const float* __restrict__ z, int D,
                           const float* __restrict__ W, const float* __restrict__ b,
                           float* __restrict__ out, int G) {
  int g = blockIdx.x * blockDim.x + threadIdx.x;
  if (g >= G) return;
  const float* zr = z + (size_t)g * D;
  float l0 = b[0], l1 = b[1];
  for (int d = 0; d < D; ++d) {
    float v = zr[d];
    l0 += v * W[d * 2 + 0];
    l1 += v * W[d * 2 + 1];
  }
  float m   = fmaxf(l0, l1);
  float lse = m + logf(expf(l0 - m) + expf(l1 - m));
  out[(size_t)g * 2 + 0] = l0 - lse;
  out[(size_t)g * 2 + 1] = l1 - lse;
}

// ---------------------------------------------------------------------------
// Parameter index map. Top-level inputs in setup_inputs() order:
//   0: x, 1: edge_index, 2: game_indices, 3..: params leaves.
// Params assumed flattened JAX-style (nested dicts, sorted keys):
//   bn dicts -> beta,gamma,mean,var ; lin -> W,b
//   top keys: bn1,bn2,bn3,conv1..3,goes_ot_head,goes_so_head,input_bn,
//             input_linear,ot_head,reg_head,so_head
//   small head keys: bn1,l1,l2,l3 ; big head keys: bn1,bn2,l1,l2,l3
// ---------------------------------------------------------------------------
enum {
  BN1_BETA = 3,                 // 3..6   (beta,gamma,mean,var)
  BN2_BETA = 7,                 // 7..10
  BN3_BETA = 11,                // 11..14
  C1_W = 15, C1_B = 16,
  C2_W = 17, C2_B = 18,
  C3_W = 19, C3_B = 20,
  GOT_BN1 = 21,                 // 21..24
  GOT_BN2 = 25,                 // 25..28
  GOT_L1W = 29, GOT_L1B = 30, GOT_L2W = 31, GOT_L2B = 32, GOT_L3W = 33, GOT_L3B = 34,
  GSO_BN1 = 35,                 // 35..38
  GSO_BN2 = 39,                 // 39..42
  GSO_L1W = 43, GSO_L1B = 44, GSO_L2W = 45, GSO_L2B = 46, GSO_L3W = 47, GSO_L3B = 48,
  IN_BN = 49,                   // 49..52
  IN_W = 53, IN_B = 54,
  OT_BN1 = 55,                  // 55..58
  OT_L1W = 59, OT_L1B = 60, OT_L2W = 61, OT_L2B = 62, OT_L3W = 63, OT_L3B = 64,
  REG_BN1 = 65,                 // 65..68
  REG_L1W = 69, REG_L1B = 70, REG_L2W = 71, REG_L2B = 72, REG_L3W = 73, REG_L3B = 74,
  SO_BN1 = 75,                  // 75..78
  SO_L1W = 79, SO_L1B = 80, SO_L2W = 81, SO_L2B = 82, SO_L3W = 83, SO_L3B = 84,
};

extern "C" void kernel_launch(void* const* d_in, const int* in_sizes, int n_in,
                              void* d_out, int out_size, void* d_ws, size_t ws_size,
                              hipStream_t stream) {
  (void)n_in; (void)out_size; (void)ws_size;

  const float* x    = (const float*)d_in[0];
  const int*   edge = (const int*)d_in[1];
  const int*   gidx = (const int*)d_in[2];
  const int NN = in_sizes[0] / 256;    // 100000
  const int E  = in_sizes[1] / 2;      // 1600000
  const int G  = in_sizes[2];          // 8192
  const int* src = edge;
  const int* dst = edge + E;

  auto P = [&](int i) -> const float* { return (const float*)d_in[i]; };

  // ---- workspace carving ----
  char* wsb = (char*)d_ws;
  size_t off = 0;
  auto alloc = [&](size_t bytes) -> void* {
    void* p = wsb + off;
    off += (bytes + 255) & ~(size_t)255;
    return p;
  };
  float*  h    = (float*)alloc((size_t)NN * 128 * 4);
  float*  hw   = (float*)alloc((size_t)NN * 128 * 4);
  float*  acc  = (float*)alloc((size_t)NN * 128 * 4);
  float*  dis  = (float*)alloc((size_t)NN * 4);
  float*  gbuf = (float*)alloc((size_t)G * 128 * 4);
  float*  z1   = (float*)alloc((size_t)G * 128 * 4);
  float*  z2   = (float*)alloc((size_t)G * 64 * 4);
  __bf16* wb   = (__bf16*)alloc((size_t)256 * 128 * 2);

  auto gemm = [&](const float* A, const float* Wf, const float* bias,
                  const float* bnG, const float* bnB, const float* bnM, const float* bnV,
                  float* out, int M, int K, int N, int flags) {
    pack_b<<<(K * N + 255) / 256, 256, 0, stream>>>(Wf, wb, K, N);
    dim3 grid((M + 127) / 128, N / 32);
    gemm_wmma_bf16<<<grid, 256, 0, stream>>>(A, wb, bias, bnG, bnB, bnM, bnV,
                                             out, M, K, N, flags);
  };

  // ---- degree / normalization ----
  zero_f32<<<(NN + 255) / 256, 256, 0, stream>>>(dis, NN);
  deg_accum<<<(E + 255) / 256, 256, 0, stream>>>(dst, dis, E);
  dis_finalize<<<(NN + 255) / 256, 256, 0, stream>>>(dis, NN);

  // ---- input layer: h = relu(bn(x @ W_in + b)) ----
  gemm(x, P(IN_W), P(IN_B),
       P(IN_BN + 1), P(IN_BN + 0), P(IN_BN + 2), P(IN_BN + 3),
       h, NN, 256, 128, EP_BIAS | EP_BN | EP_RELU);

  // ---- 3 GCN layers ----
  const int cw[3]  = {C1_W, C2_W, C3_W};
  const int cb[3]  = {C1_B, C2_B, C3_B};
  const int cbn[3] = {BN1_BETA, BN2_BETA, BN3_BETA};
  const long accN = (long)NN * 128;
  for (int l = 0; l < 3; ++l) {
    gemm(h, P(cw[l]), nullptr, nullptr, nullptr, nullptr, nullptr,
         hw, NN, 128, 128, 0);
    zero_f32<<<(int)((accN + 255) / 256), 256, 0, stream>>>(acc, accN);
    long st = (long)E * 32;
    edge_scatter<<<(int)((st + 255) / 256), 256, 0, stream>>>(hw, src, dst, dis, acc, E);
    gcn_finalize<<<(int)((accN + 255) / 256), 256, 0, stream>>>(
        h, hw, acc, dis, P(cb[l]),
        P(cbn[l] + 1), P(cbn[l] + 0), P(cbn[l] + 2), P(cbn[l] + 3), NN);
  }

  // ---- gather game rows ----
  gather_rows<<<(G * 128 + 255) / 256, 256, 0, stream>>>(h, gidx, gbuf, G);

  // ---- heads (output order: reg, ot, so, goes_ot, goes_so) ----
  float* out = (float*)d_out;

  auto small_head = [&](int bn1, int l1w, int l1b, int l2w, int l2b,
                        int l3w, int l3b, float* o) {
    gemm(gbuf, P(l1w), P(l1b),
         P(bn1 + 1), P(bn1 + 0), P(bn1 + 2), P(bn1 + 3),
         z1, G, 128, 64, EP_BIAS | EP_BN | EP_RELU);
    gemm(z1, P(l2w), P(l2b), nullptr, nullptr, nullptr, nullptr,
         z2, G, 64, 32, EP_BIAS | EP_RELU);
    head_final<<<(G + 255) / 256, 256, 0, stream>>>(z2, 32, P(l3w), P(l3b), o, G);
  };
  auto big_head = [&](int bn1, int bn2, int l1w, int l1b, int l2w, int l2b,
                      int l3w, int l3b, float* o) {
    gemm(gbuf, P(l1w), P(l1b),
         P(bn1 + 1), P(bn1 + 0), P(bn1 + 2), P(bn1 + 3),
         z1, G, 128, 128, EP_BIAS | EP_BN | EP_RELU);
    gemm(z1, P(l2w), P(l2b),
         P(bn2 + 1), P(bn2 + 0), P(bn2 + 2), P(bn2 + 3),
         z2, G, 128, 64, EP_BIAS | EP_BN | EP_RELU);
    head_final<<<(G + 255) / 256, 256, 0, stream>>>(z2, 64, P(l3w), P(l3b), o, G);
  };

  small_head(REG_BN1, REG_L1W, REG_L1B, REG_L2W, REG_L2B, REG_L3W, REG_L3B, out + 0L * G * 2);
  small_head(OT_BN1,  OT_L1W,  OT_L1B,  OT_L2W,  OT_L2B,  OT_L3W,  OT_L3B,  out + 1L * G * 2);
  small_head(SO_BN1,  SO_L1W,  SO_L1B,  SO_L2W,  SO_L2B,  SO_L3W,  SO_L3B,  out + 2L * G * 2);
  big_head(GOT_BN1, GOT_BN2, GOT_L1W, GOT_L1B, GOT_L2W, GOT_L2B, GOT_L3W, GOT_L3B, out + 3L * G * 2);
  big_head(GSO_BN1, GSO_BN2, GSO_L1W, GSO_L1B, GSO_L2W, GSO_L2B, GSO_L3W, GSO_L3B, out + 4L * G * 2);
}
